// GIN_936302870559
// MI455X (gfx1250) — matrix operand order repaired and checked by reading, not compiled, not run on hardware
//
#include <hip/hip_runtime.h>
#include <stdint.h>

#define N_NODES 100000
#define N_EDGES 600000
#define BATCH   20000
#define IN_DIM  64
#define HID     128
#define OUT_DIM 10

typedef __attribute__((ext_vector_type(16))) __bf16 bf16x16;
typedef __attribute__((ext_vector_type(8)))  float  f32x8;

union Frag {
    uint32_t u[8];
    uint4    q[2];
    __bf16   h[16];
    bf16x16  v;
};
static_assert(sizeof(Frag) == 32, "frag size");

union Pk2 {
    __bf16   h[2];
    uint32_t u;
};

// ---------------------------------------------------------------------------
// Pack an fp32 (K x 128) weight matrix into bf16 WMMA B-fragment order.
// dword index = ((nt*KT + kt)*32 + lane)*8 + r
// B layout (16-bit, 32x16 tile): K = kt*32 + (lane>>4)*8 + 2(r&3){,+1} + 16*(r>>2)
//                                (matches A: V0-3 -> hi*8+0..7, V4-7 -> 16+hi*8+0..7)
// n = nt*16 + (lane & 15)
// ---------------------------------------------------------------------------
__global__ void gin_pack_b(const float* __restrict__ w, uint32_t* __restrict__ wp, int K) {
    int total = K * 64;                         // K*128 bf16 -> K*64 dwords
    int idx = blockIdx.x * blockDim.x + threadIdx.x;
    if (idx >= total) return;
    int KT   = K >> 5;
    int r    = idx & 7;
    int lane = (idx >> 3) & 31;
    int rest = idx >> 8;
    int kt   = rest % KT;
    int nt   = rest / KT;
    int n    = nt * 16 + (lane & 15);
    int k    = kt * 32 + (lane >> 4) * 8 + (r & 3) * 2 + (r >> 2) * 16;
    Pk2 p;
    p.h[0] = (__bf16)w[(size_t)k * HID + n];
    p.h[1] = (__bf16)w[(size_t)(k + 1) * HID + n];
    wp[idx] = p.u;
}

// ---------------------------------------------------------------------------
// z = h (fp32 vector copy, float4)
// ---------------------------------------------------------------------------
__global__ void gin_copy4(const float* __restrict__ src, float* __restrict__ dst, int n4) {
    int i = blockIdx.x * blockDim.x + threadIdx.x;
    if (i < n4) ((float4*)dst)[i] = ((const float4*)src)[i];
}

// ---------------------------------------------------------------------------
// z[dst] += h[src] over edges; one wave per edge, lane-strided columns.
// (h fits in the 192MB L2, so these atomics are L2-resident.)
// ---------------------------------------------------------------------------
__global__ void gin_scatter(const float* __restrict__ h, const long long* __restrict__ ei,
                            float* __restrict__ z, int dim) {
    int e = blockIdx.x * (blockDim.x >> 5) + (threadIdx.x >> 5);
    if (e >= N_EDGES) return;
    int lane = threadIdx.x & 31;
    long long s = ei[e];
    long long d = ei[N_EDGES + e];
    const float* hs = h + (size_t)s * dim;
    float*       zd = z + (size_t)d * dim;
    for (int c = lane; c < dim; c += 32)
        atomicAdd(&zd[c], hs[c]);
}

// ---------------------------------------------------------------------------
// Fused GIN MLP: hout = ReLU( ReLU(z @ wa + ba) @ wb + bb )
// z: (nrows x K) fp32.  waPk: bf16 packed (K x 128).  wbPk: bf16 packed (128 x 128).
// 8 waves/block, 16 rows/wave -> 128 rows/block.  v_wmma_f32_16x16x32_bf16.
// ---------------------------------------------------------------------------
template <int K>
__global__ __launch_bounds__(256) void gin_mlp(const float* __restrict__ z,
                                               const uint32_t* __restrict__ waPk,
                                               const float* __restrict__ ba,
                                               const uint32_t* __restrict__ wbPk,
                                               const float* __restrict__ bb,
                                               float* __restrict__ hout, int nrows) {
    constexpr int KT1 = K / 32;
    __shared__ uint32_t lds_w[128 * 64];     // 32 KB: holds wa, then wb
    __shared__ uint32_t lds_t[8][1024];      // 32 KB: per-wave 16x128 bf16 intermediate

    const int tid  = threadIdx.x;
    const int wave = tid >> 5;
    const int lane = tid & 31;
    const int ln   = lane & 15;              // A/B/C "low" index within half
    const int hi   = lane >> 4;              // lane-half select
    const int row0 = blockIdx.x * 128 + wave * 16;

    // ---- stage wa in LDS ----
    for (int i = tid; i < K * 64; i += 256) lds_w[i] = waPk[i];
    __syncthreads();

    // ---- GEMM1: acc = z @ wa ----
    // Out-of-range rows: clamp the address (no branches). Their garbage stays in
    // their own output row, which is discarded by the guarded final store.
    const int rowA = row0 + ln;
    int       rowC = rowA < nrows ? rowA : (nrows - 1);
    const float* zr = z + (size_t)rowC * K + hi * 8;

    f32x8 acc[8];
#pragma unroll
    for (int nt = 0; nt < 8; ++nt) acc[nt] = (f32x8){0, 0, 0, 0, 0, 0, 0, 0};

#pragma unroll
    for (int kt = 0; kt < KT1; ++kt) {
        // 16-bit A layout: V0-3 -> K = hi*8 + 0..7 ; V4-7 -> K = 16 + hi*8 + 0..7
        Frag a;
        float4 f0 = *(const float4*)(zr + kt * 32);
        float4 f1 = *(const float4*)(zr + kt * 32 + 4);
        float4 f2 = *(const float4*)(zr + kt * 32 + 16);
        float4 f3 = *(const float4*)(zr + kt * 32 + 20);
        a.h[0]  = (__bf16)f0.x;  a.h[1]  = (__bf16)f0.y;
        a.h[2]  = (__bf16)f0.z;  a.h[3]  = (__bf16)f0.w;
        a.h[4]  = (__bf16)f1.x;  a.h[5]  = (__bf16)f1.y;
        a.h[6]  = (__bf16)f1.z;  a.h[7]  = (__bf16)f1.w;
        a.h[8]  = (__bf16)f2.x;  a.h[9]  = (__bf16)f2.y;
        a.h[10] = (__bf16)f2.z;  a.h[11] = (__bf16)f2.w;
        a.h[12] = (__bf16)f3.x;  a.h[13] = (__bf16)f3.y;
        a.h[14] = (__bf16)f3.z;  a.h[15] = (__bf16)f3.w;
#pragma unroll
        for (int nt = 0; nt < 8; ++nt) {
            Frag b;
            const uint4* bp = (const uint4*)&lds_w[(((nt * KT1) + kt) * 32 + lane) * 8];
            b.q[0] = bp[0];
            b.q[1] = bp[1];
            acc[nt] = __builtin_amdgcn_wmma_f32_16x16x32_bf16(
                false, a.v, false, b.v, (short)0, acc[nt], false, false);
        }
    }

    // ---- bias + ReLU + bf16 transpose-stage to LDS (C layout -> row-major) ----
    __bf16* trow = (__bf16*)lds_t[wave];
#pragma unroll
    for (int nt = 0; nt < 8; ++nt) {
        const int   n  = nt * 16 + ln;
        const float bv = ba[n];
#pragma unroll
        for (int r = 0; r < 8; ++r) {
            const int mm = r + hi * 8;               // C layout: M = r (+8 for hi lanes)
            float v = acc[nt][r] + bv;
            v = v > 0.f ? v : 0.f;
            trow[mm * 128 + n] = (__bf16)v;
        }
    }

    // ---- swap weights: wb into LDS ----
    __syncthreads();
    for (int i = tid; i < 8192; i += 256) lds_w[i] = wbPk[i];
    __syncthreads();

    // ---- GEMM2: acc2 = t @ wb ----
    f32x8 acc2[8];
#pragma unroll
    for (int nt = 0; nt < 8; ++nt) acc2[nt] = (f32x8){0, 0, 0, 0, 0, 0, 0, 0};

    const uint32_t* tread = lds_t[wave] + ln * 64;   // row ln of t, dword (K-pair) indexed
#pragma unroll
    for (int kt = 0; kt < 4; ++kt) {
        Frag a;
#pragma unroll
        for (int r = 0; r < 8; ++r) {
            int k = kt * 32 + ((r < 4) ? (hi * 8 + 2 * r) : (16 + hi * 8 + 2 * (r - 4)));
            a.u[r] = tread[k >> 1];                  // k even -> aligned dword (2 bf16)
        }
#pragma unroll
        for (int nt = 0; nt < 8; ++nt) {
            Frag b;
            const uint4* bp = (const uint4*)&lds_w[(((nt * 4) + kt) * 32 + lane) * 8];
            b.q[0] = bp[0];
            b.q[1] = bp[1];
            acc2[nt] = __builtin_amdgcn_wmma_f32_16x16x32_bf16(
                false, a.v, false, b.v, (short)0, acc2[nt], false, false);
        }
    }

    // ---- bias + ReLU + guarded fp32 store ----
#pragma unroll
    for (int nt = 0; nt < 8; ++nt) {
        const int   n  = nt * 16 + ln;
        const float bv = bb[n];
#pragma unroll
        for (int r = 0; r < 8; ++r) {
            const int mm  = r + hi * 8;
            const int row = row0 + mm;
            if (row < nrows) {
                float v = acc2[nt][r] + bv;
                hout[(size_t)row * HID + n] = v > 0.f ? v : 0.f;
            }
        }
    }
}

// ---------------------------------------------------------------------------
// Final FC: (20000 x 640) @ (640 x 10) + b  (tiny; one thread per output)
// ---------------------------------------------------------------------------
__global__ void gin_fc(const float* __restrict__ h, const float* __restrict__ wfc,
                       const float* __restrict__ bfc, float* __restrict__ out) {
    int idx = blockIdx.x * blockDim.x + threadIdx.x;
    if (idx >= BATCH * OUT_DIM) return;
    int g = idx / OUT_DIM;
    int o = idx % OUT_DIM;
    const float* hr = h + (size_t)g * (5 * HID);
    float acc = bfc[o];
#pragma unroll 4
    for (int j = 0; j < 5 * HID; ++j) acc = fmaf(hr[j], wfc[j * OUT_DIM + o], acc);
    out[idx] = acc;
}

// ---------------------------------------------------------------------------
extern "C" void kernel_launch(void* const* d_in, const int* in_sizes, int n_in,
                              void* d_out, int out_size, void* d_ws, size_t ws_size,
                              hipStream_t stream) {
    const float*     x   = (const float*)d_in[0];
    const long long* ei  = (const long long*)d_in[1];
    const float*     w1a = (const float*)d_in[2];
    const float*     b1a = (const float*)d_in[3];
    const float*     w1b = (const float*)d_in[4];
    const float*     b1b = (const float*)d_in[5];
    const float*     wsa = (const float*)d_in[6];
    const float*     bsa = (const float*)d_in[7];
    const float*     wsb = (const float*)d_in[8];
    const float*     bsb = (const float*)d_in[9];
    const float*     wfc = (const float*)d_in[10];
    const float*     bfc = (const float*)d_in[11];
    float*           out = (float*)d_out;

    // workspace layout
    char*  ws   = (char*)d_ws;
    float* zbuf = (float*)ws;                                        // N*128 fp32
    float* hbuf = (float*)(ws + (size_t)N_NODES * HID * sizeof(float));
    uint32_t* pk    = (uint32_t*)(ws + 2 * (size_t)N_NODES * HID * sizeof(float));
    uint32_t* w1aPk = pk;                    // 64*64  = 4096 dwords
    uint32_t* w1bPk = pk + 4096;             // 128*64 = 8192 dwords
    uint32_t* wsaPk = w1bPk + 8192;          // 4 * 8192
    uint32_t* wsbPk = wsaPk + 4 * 8192;      // 4 * 8192

    // --- pack all weight matrices to bf16 WMMA-B layout ---
    gin_pack_b<<<(4096 + 255) / 256, 256, 0, stream>>>(w1a, w1aPk, IN_DIM);
    gin_pack_b<<<32, 256, 0, stream>>>(w1b, w1bPk, HID);
    for (int i = 0; i < 4; ++i) {
        gin_pack_b<<<32, 256, 0, stream>>>(wsa + (size_t)i * HID * HID, wsaPk + i * 8192, HID);
        gin_pack_b<<<32, 256, 0, stream>>>(wsb + (size_t)i * HID * HID, wsbPk + i * 8192, HID);
    }

    const int edgeBlocks = (N_EDGES + 7) / 8;           // 8 edges (waves) per block
    const int mlpBlocks  = (N_NODES + 127) / 128;

    // --- layer 1 (K = 64) ---
    gin_copy4<<<(N_NODES * IN_DIM / 4 + 255) / 256, 256, 0, stream>>>(x, zbuf, N_NODES * IN_DIM / 4);
    gin_scatter<<<edgeBlocks, 256, 0, stream>>>(x, ei, zbuf, IN_DIM);
    gin_mlp<IN_DIM><<<mlpBlocks, 256, 0, stream>>>(zbuf, w1aPk, b1a, w1bPk, b1b, hbuf, N_NODES);

    // --- layers 2..5 (K = 128) ---
    for (int i = 0; i < 4; ++i) {
        gin_copy4<<<(N_NODES * HID / 4 + 255) / 256, 256, 0, stream>>>(hbuf, zbuf, N_NODES * HID / 4);
        gin_scatter<<<edgeBlocks, 256, 0, stream>>>(hbuf, ei, zbuf, HID);
        gin_mlp<HID><<<mlpBlocks, 256, 0, stream>>>(zbuf, wsaPk + i * 8192, bsa + i * HID,
                                                    wsbPk + i * 8192, bsb + i * HID, hbuf, N_NODES);
    }

    // --- readout FC ---
    gin_fc<<<(BATCH * OUT_DIM + 255) / 256, 256, 0, stream>>>(hbuf, wfc, bfc, out);
}
